// ScaledDotProductAttention_63625645523035
// MI455X (gfx1250) — matrix operand order repaired and checked by reading, not compiled, not run on hardware
//
#include <hip/hip_runtime.h>
#include <hip/hip_bf16.h>

typedef __attribute__((ext_vector_type(16))) _Float16 v16h;
typedef __attribute__((ext_vector_type(8)))  float    v8f;

#define S_LEN 2048
#define D_DIM 128
#define QT 128          // queries per block
#define KT 64           // keys per tile
#define NW 4            // waves per block (wave32), each wave owns 32 query rows
#define KS 136          // LDS stride for K tile (padded, even)
#define VS 72           // LDS stride for V^T tile (padded, even)

union AFrag { v16h v; unsigned u[8]; _Float16 h[16]; };

__device__ __forceinline__ unsigned pack2h(float a, float b) {
    union { _Float16 h[2]; unsigned u; } t;
    t.h[0] = (_Float16)a; t.h[1] = (_Float16)b;
    return t.u;
}

// DPP row-rotate (16-lane rows): VALU-pipe cross-lane, keeps lanes 0-15 / 16-31 separate.
template <int CTRL>
__device__ __forceinline__ float dpp_rot(float v) {
    return __int_as_float(__builtin_amdgcn_mov_dpp(__float_as_int(v), CTRL, 0xf, 0xf, true));
}
// row_ror:8 = 0x128, row_ror:4 = 0x124, row_ror:2 = 0x122, row_ror:1 = 0x121
__device__ __forceinline__ float rowmax16(float v) {
    v = fmaxf(v, dpp_rot<0x128>(v));
    v = fmaxf(v, dpp_rot<0x124>(v));
    v = fmaxf(v, dpp_rot<0x122>(v));
    v = fmaxf(v, dpp_rot<0x121>(v));
    return v;
}
__device__ __forceinline__ float rowsum16(float v) {
    v += dpp_rot<0x128>(v);
    v += dpp_rot<0x124>(v);
    v += dpp_rot<0x122>(v);
    v += dpp_rot<0x121>(v);
    return v;
}

__global__ __launch_bounds__(128, 1)
void fa_causal_kernel(const float* __restrict__ Q, const float* __restrict__ K,
                      const float* __restrict__ V, const int* __restrict__ pmask,
                      float* __restrict__ Out) {
    __shared__ _Float16 sK[KT * KS];         // K tile, row-major [key][dim]
    __shared__ _Float16 sVT[D_DIM * VS];     // V tile transposed [dim][key]
    __shared__ _Float16 sP[NW * 32 * KT];    // per-wave probability tile 32x64

    const int tid  = threadIdx.x;
    const int w    = tid >> 5;
    const int lane = tid & 31;
    const int hg   = lane >> 4;    // half-group (0 or 1)
    const int ln   = lane & 15;
    const int b    = blockIdx.y;
    const int q0   = blockIdx.x * QT;
    const int qw   = q0 + w * 32;               // first query row of this wave
    // 1/sqrt(128) * log2(e): softmax runs in base-2 domain (hardware exp is base-2)
    const float scale = 0.12751691579231642f;
    const float NEGINF = -__builtin_inff();

    // ---- Preload Q A-fragments for both stripes (scaled, f32 -> f16) ----
    AFrag qa[2][4];
#pragma unroll
    for (int s = 0; s < 2; ++s) {
        const float* qbase = Q + ((size_t)b * S_LEN + qw + s * 16 + ln) * D_DIM;
#pragma unroll
        for (int c = 0; c < 4; ++c)
#pragma unroll
            for (int r = 0; r < 8; ++r) {
                int k0 = c * 32 + 2 * r + hg * 16;
                qa[s][c].u[r] = pack2h(qbase[k0] * scale, qbase[k0 + 1] * scale);
            }
    }

    // ---- Online-softmax state (per stripe), in log2 domain ----
    float m[2][8], l[2][8];
    v8f o[2][8];
#pragma unroll
    for (int s = 0; s < 2; ++s)
#pragma unroll
        for (int r = 0; r < 8; ++r) { m[s][r] = -1e30f; l[s][r] = 0.f; }
#pragma unroll
    for (int s = 0; s < 2; ++s)
#pragma unroll
        for (int n = 0; n < 8; ++n)
            o[s][n] = (v8f){0.f,0.f,0.f,0.f,0.f,0.f,0.f,0.f};

    const int nkt = (q0 + QT) / KT;   // causal: only key tiles <= last query
    for (int kt = 0; kt < nkt; ++kt) {
        const int kb = kt * KT;
        __syncthreads();
        // ---- Stage K tile (f16 row-major) and V tile (f16 transposed) ----
#pragma unroll
        for (int i = 0; i < 16; ++i) {
            int e4  = tid + i * 128;          // 2048 float4s = 64x128 floats
            int row = e4 >> 5;
            int c4  = (e4 & 31) * 4;
            float4 kv = *(const float4*)(K + ((size_t)b * S_LEN + kb + row) * D_DIM + c4);
            _Float16* kd = &sK[row * KS + c4];
            *(unsigned*)&kd[0] = pack2h(kv.x, kv.y);
            *(unsigned*)&kd[2] = pack2h(kv.z, kv.w);
            float4 vv = *(const float4*)(V + ((size_t)b * S_LEN + kb + row) * D_DIM + c4);
            sVT[(c4 + 0) * VS + row] = (_Float16)vv.x;
            sVT[(c4 + 1) * VS + row] = (_Float16)vv.y;
            sVT[(c4 + 2) * VS + row] = (_Float16)vv.z;
            sVT[(c4 + 3) * VS + row] = (_Float16)vv.w;
        }
        __syncthreads();

        // ---- Prefetch next tile's K/V cachelines (global_prefetch_b8: no VGPR,
        //      no counter; warms WGP$/L2 while this tile computes) ----
        if (kt + 1 < nkt) {
            const int kb2 = kb + KT;
#pragma unroll
            for (int i = 0; i < 2; ++i) {
                int li  = tid + i * 128;          // 256 cachelines of 128B per matrix
                int row = li >> 2;
                int col = (li & 3) * 32;
                __builtin_prefetch(K + ((size_t)b * S_LEN + kb2 + row) * D_DIM + col, 0, 3);
                __builtin_prefetch(V + ((size_t)b * S_LEN + kb2 + row) * D_DIM + col, 0, 3);
            }
        }

        // Wave-uniform: this wave's rows [qw, qw+31] see nothing of a fully-future tile
        if (kb > qw + 31) continue;

        // ---- S = (Q*scale) K^T : each B fragment feeds BOTH stripes ----
        v8f sf[2][4];
#pragma unroll
        for (int s = 0; s < 2; ++s)
#pragma unroll
            for (int f = 0; f < 4; ++f)
                sf[s][f] = (v8f){0.f,0.f,0.f,0.f,0.f,0.f,0.f,0.f};
#pragma unroll
        for (int f = 0; f < 4; ++f) {
#pragma unroll
            for (int c = 0; c < 4; ++c) {
                AFrag bf;
                const _Float16* kr = &sK[(f * 16 + ln) * KS + c * 32 + hg * 16];
#pragma unroll
                for (int r = 0; r < 8; ++r) bf.u[r] = *(const unsigned*)&kr[2 * r];
                sf[0][f] = __builtin_amdgcn_wmma_f32_16x16x32_f16(
                             false, qa[0][c].v, false, bf.v, (short)0, sf[0][f], false, false);
                sf[1][f] = __builtin_amdgcn_wmma_f32_16x16x32_f16(
                             false, qa[1][c].v, false, bf.v, (short)0, sf[1][f], false, false);
            }
        }

        // ---- Column bias from padding mask (0 or -inf), per N-tile ----
        float cb[4];
#pragma unroll
        for (int f = 0; f < 4; ++f)
            cb[f] = (pmask[(size_t)b * S_LEN + kb + f * 16 + ln] != 0) ? 0.f : NEGINF;

        _Float16* pw = &sP[w * 32 * KT];
#pragma unroll
        for (int s = 0; s < 2; ++s) {
            float nm[8];
#pragma unroll
            for (int r = 0; r < 8; ++r) nm[r] = m[s][r];

            // Causal compares only needed when the tile crosses this stripe's diagonal
            if (kb + KT - 1 > qw + s * 16) {
#pragma unroll
                for (int f = 0; f < 4; ++f) {
                    int kg = kb + f * 16 + ln;
#pragma unroll
                    for (int r = 0; r < 8; ++r) {
                        int qg = qw + s * 16 + r + hg * 8;
                        float sv = (kg <= qg) ? (sf[s][f][r] + cb[f]) : NEGINF;
                        sf[s][f][r] = sv;
                        nm[r] = fmaxf(nm[r], sv);
                    }
                }
            } else {
#pragma unroll
                for (int f = 0; f < 4; ++f)
#pragma unroll
                    for (int r = 0; r < 8; ++r) {
                        float sv = sf[s][f][r] + cb[f];
                        sf[s][f][r] = sv;
                        nm[r] = fmaxf(nm[r], sv);
                    }
            }
#pragma unroll
            for (int r = 0; r < 8; ++r) nm[r] = rowmax16(nm[r]);   // DPP, VALU pipe

            float alpha[8], rs[8];
#pragma unroll
            for (int r = 0; r < 8; ++r) {
                alpha[r] = __builtin_amdgcn_exp2f(m[s][r] - nm[r]);
                m[s][r] = nm[r];
            }
#pragma unroll
            for (int f = 0; f < 4; ++f)
#pragma unroll
                for (int r = 0; r < 8; ++r) {
                    float p = __builtin_amdgcn_exp2f(sf[s][f][r] - nm[r]);
                    sf[s][f][r] = p;
                    if (f == 0) rs[r] = p; else rs[r] += p;
                }
#pragma unroll
            for (int r = 0; r < 8; ++r) rs[r] = rowsum16(rs[r]);   // DPP, VALU pipe
#pragma unroll
            for (int r = 0; r < 8; ++r) l[s][r] = l[s][r] * alpha[r] + rs[r];
#pragma unroll
            for (int n = 0; n < 8; ++n)
#pragma unroll
                for (int r = 0; r < 8; ++r) o[s][n][r] *= alpha[r];

            // C-layout P -> per-wave LDS (DS ops in-order within the wave)
#pragma unroll
            for (int f = 0; f < 4; ++f)
#pragma unroll
                for (int r = 0; r < 8; ++r)
                    pw[(s * 16 + r + hg * 8) * KT + f * 16 + ln] = (_Float16)sf[s][f][r];
        }

        // ---- O += P V : each V B-fragment feeds BOTH stripes ----
#pragma unroll
        for (int kc = 0; kc < 2; ++kc) {
            AFrag pa[2];
#pragma unroll
            for (int s = 0; s < 2; ++s) {
                const _Float16* pr = &pw[(s * 16 + ln) * KT + kc * 32 + hg * 16];
#pragma unroll
                for (int r = 0; r < 8; ++r) {
                    int off = 2 * r;
                    pa[s].u[r] = *(const unsigned*)&pr[off];
                }
            }
#pragma unroll
            for (int n = 0; n < 8; ++n) {
                AFrag vb;
                const _Float16* vr = &sVT[(n * 16 + ln) * VS + kc * 32 + hg * 16];
#pragma unroll
                for (int r = 0; r < 8; ++r) vb.u[r] = *(const unsigned*)&vr[2 * r];
                o[0][n] = __builtin_amdgcn_wmma_f32_16x16x32_f16(
                            false, pa[0].v, false, vb.v, (short)0, o[0][n], false, false);
                o[1][n] = __builtin_amdgcn_wmma_f32_16x16x32_f16(
                            false, pa[1].v, false, vb.v, (short)0, o[1][n], false, false);
            }
        }
    }

    // ---- Normalize and write O (fp32) ----
#pragma unroll
    for (int s = 0; s < 2; ++s)
#pragma unroll
        for (int n = 0; n < 8; ++n)
#pragma unroll
            for (int r = 0; r < 8; ++r) {
                int qg = qw + s * 16 + r + hg * 8;
                float inv = 1.0f / l[s][r];
                Out[((size_t)b * S_LEN + qg) * D_DIM + n * 16 + ln] = o[s][n][r] * inv;
            }
}

extern "C" void kernel_launch(void* const* d_in, const int* in_sizes, int n_in,
                              void* d_out, int out_size, void* d_ws, size_t ws_size,
                              hipStream_t stream) {
    const float* Q = (const float*)d_in[0];
    const float* K = (const float*)d_in[1];
    const float* V = (const float*)d_in[2];
    const int* pmask = (const int*)d_in[3];
    // d_in[4] = sequence_mask (causal tril) -- applied analytically in-kernel
    float* Out = (float*)d_out;
    dim3 grid(S_LEN / QT, 16 /*B*/);
    dim3 block(128);
    fa_causal_kernel<<<grid, block, 0, stream>>>(Q, K, V, pmask, Out);
}
